// ERPDistortionAwareDCN_62414464745681
// MI455X (gfx1250) — compile-verified, matrix operand order checked by reading, not run on hardware
//
#include <hip/hip_runtime.h>
#include <hip/hip_bf16.h>

// ---------------------------------------------------------------------------
// Problem constants
// ---------------------------------------------------------------------------
#define BB 2
#define CC 128
#define HH 128
#define WW 256
#define HWX (HH*WW)          // 32768
#define AUXC 133
#define AUXP 144             // padded aux channels (zeros 133..143)
#define KOG 1296             // 9*AUXP
#define KPOG 1312            // padded to 41*32
#define KCOG 41
#define KB 1152              // 9*128
#define KCB 36
#define MT_OG 6              // 96 padded out channels (18 offset + 66 gate + pad)
#define MT_C 8               // 128 out channels

// raw TDM tile geometry: 3 rows x 18 pixels x channels
#define SP_PXS 136           // base tile pixel stride in u16 (128 + 8 pad = 68 dwords)
#define SP_SZ  (3*18*SP_PXS) // 7344 u16
#define OG_SZ  (3*18*AUXP)   // 7776 u16 (no pad)

typedef __attribute__((ext_vector_type(16))) __bf16 bf16x16;
typedef __attribute__((ext_vector_type(8)))  float  f32x8;
typedef __attribute__((ext_vector_type(8)))  int    i32x8;
typedef __attribute__((ext_vector_type(4)))  int    i32x4;
typedef __attribute__((ext_vector_type(4)))  unsigned u32x4;

#if defined(__gfx1250__) && __has_builtin(__builtin_amdgcn_tensor_load_to_lds) && \
    __has_builtin(__builtin_amdgcn_s_wait_tensorcnt)
#define USE_TDM 1
#else
#define USE_TDM 0
#endif

// ---------------------------------------------------------------------------
// Helpers
// ---------------------------------------------------------------------------
__device__ __forceinline__ unsigned short f2bf(float f) {
  unsigned u = __float_as_uint(f);
  unsigned r = u + 0x7fffu + ((u >> 16) & 1u);
  if ((u & 0x7f800000u) == 0x7f800000u) r = u;   // inf/nan passthrough
  return (unsigned short)(r >> 16);
}

__device__ __forceinline__ bf16x16 frag_from(uint4 a, uint4 b) {
  i32x8 t = { (int)a.x,(int)a.y,(int)a.z,(int)a.w,
              (int)b.x,(int)b.y,(int)b.z,(int)b.w };
  return __builtin_bit_cast(bf16x16, t);
}

// A-fragment: packed so each lane's 16 halves are contiguous (32B)
__device__ __forceinline__ bf16x16 load_a_frag(const unsigned short* W, int block, int lane) {
  const uint4* p = (const uint4*)W + (size_t)block * 64 + lane * 2;
  return frag_from(p[0], p[1]);
}

// B-fragment from a per-pixel-contiguous-K LDS row (used for deform patch)
__device__ __forceinline__ bf16x16 load_b_frag(const unsigned short* pixrow, int kbase, int lane) {
  const uint4* p = (const uint4*)(pixrow + kbase + ((lane & 16) ? 8 : 0));
  return frag_from(p[0], p[2]);
}

// 8-half run offset inside raw base tile, relative to (Sp + pix*SP_PXS).
// k = tap*128 + c ; constant-folds when k is an unroll constant.
__device__ __forceinline__ const uint4* sp_run(const unsigned short* pbase, int k) {
  int tap = k >> 7, c = k & 127;
  int row = tap / 3, dx = tap - row * 3;
  return (const uint4*)(pbase + ((row * 18) + dx) * SP_PXS + c);
}

// 8-half run offset inside raw aux tile, relative to (Ar + pix*AUXP).
// k = tap*144 + c ; k >= KOG -> zero block (GEMM K padding 1296..1311)
__device__ __forceinline__ const uint4* og_run(const unsigned short* pbase,
                                               const unsigned short* zpad, int k) {
  int tap = k / AUXP;
  int c = k - tap * AUXP;
  int row = tap / 3, dx = tap - row * 3;
  const uint4* q = (const uint4*)(pbase + ((row * 18) + dx) * AUXP + c);
  return (k < KOG) ? q : (const uint4*)zpad;
}

#if USE_TDM
// 2D TDM tile load: tile_d0 elements per line, tile_d1 lines, stride in elements
// (data_size = 2 bytes). Optional LDS padding: 4 DWORDs every 64 DWORDs.
// 6-arg builtin form: (g0:u32x4, g1:i32x8, g2:i32x4, g3:i32x4, g4:i32x8, cpol)
__device__ __forceinline__ void tdm_load_2d(unsigned lds_off, const void* gaddr,
                                            unsigned tile_d0, unsigned tile_d1,
                                            unsigned long stride_elts, bool pad) {
  unsigned long ga = (unsigned long)gaddr;
  u32x4 g0;
  g0.x = 1u;                                                // count=1, user descriptor
  g0.y = lds_off;                                           // LDS byte address
  g0.z = (unsigned)(ga & 0xffffffffu);                      // global_addr[31:0]
  g0.w = (unsigned)((ga >> 32) & 0x01ffffffu) | (2u << 30); // global_addr[56:32], type=2
  const unsigned td0 = 0x40000000u, td1 = 0x40000000u;      // huge tensor dims (interior only)
  unsigned d0 = (1u << 16);                                 // data_size=2B, mask=0
  if (pad) d0 |= (1u << 20) | (5u << 22) | (3u << 25);      // pad_en, every 64 DW, +4 DW
  i32x8 g1;
  g1[0] = (int)d0;
  g1[1] = (int)((td0 & 0xffffu) << 16);                     // atomic_addr=0 | td0[15:0]
  g1[2] = (int)((td0 >> 16) | ((td1 & 0xffffu) << 16));     // td0[31:16] | td1[15:0]
  g1[3] = (int)((td1 >> 16) | (tile_d0 << 16));             // td1[31:16] | tile_dim0
  g1[4] = (int)(tile_d1 & 0xffffu);                         // tile_dim1 | tile_dim2=0
  g1[5] = (int)(stride_elts & 0xffffffffu);                 // dim0_stride[31:0]
  g1[6] = (int)((stride_elts >> 32) & 0xffffu);             // dim0_stride[47:32] | s1=0
  g1[7] = 0;
  i32x4 z4 = {0, 0, 0, 0};
  i32x8 z8 = {0, 0, 0, 0, 0, 0, 0, 0};
  __builtin_amdgcn_tensor_load_to_lds(g0, g1, z4, z4, z8, 0);
}
#endif

// ---------------------------------------------------------------------------
// 1) Weight packing into bf16 A-fragment order
// ---------------------------------------------------------------------------
__global__ __launch_bounds__(256) void pack_weights_kernel(
    const float* __restrict__ offw, const float* __restrict__ gw1,
    const float* __restrict__ bw,   const float* __restrict__ dw,
    unsigned short* __restrict__ Wog, unsigned short* __restrict__ Wb,
    unsigned short* __restrict__ Wd) {
  const int NOG = MT_OG * KCOG * 512;   // 125952
  const int NB  = MT_C  * KCB  * 512;   // 147456
  int idx = blockIdx.x * 256 + threadIdx.x;
  unsigned short* dst; int local, KC, kind;
  if (idx < NOG)            { dst = Wog; local = idx;            KC = KCOG; kind = 0; }
  else if (idx < NOG + NB)  { dst = Wb;  local = idx - NOG;      KC = KCB;  kind = 1; }
  else if (idx < NOG + 2*NB){ dst = Wd;  local = idx - NOG - NB; KC = KCB;  kind = 2; }
  else return;
  int block  = local >> 9;
  int within = local & 511;
  int lane = within >> 4, half = within & 15;
  int mt = block / KC, kc = block % KC;
  int m = mt * 16 + (lane & 15);
  int k = kc * 32 + (half >> 3) * 16 + ((lane & 16) ? 8 : 0) + (half & 7);
  float v = 0.0f;
  if (kind == 0) {                       // offset(18) + gate1(66), k = tap*144 + c
    if (k < KOG) {
      int tap = k / AUXP, c = k % AUXP;
      if (c < AUXC) {
        if (m < 18)      v = offw[(m * AUXC + c) * 9 + tap];
        else if (m < 84) v = gw1[((m - 18) * AUXC + c) * 9 + tap];
      }
    }
  } else if (kind == 1) {                // base conv, k = tap*128 + cin
    int tap = k >> 7, c = k & 127;
    v = bw[(m * CC + c) * 9 + tap];
  } else {                               // def conv, k = cin*9 + tap
    int c = k / 9, tap = k - c * 9;
    v = dw[(m * CC + c) * 9 + tap];
  }
  dst[local] = f2bf(v);
}

// ---------------------------------------------------------------------------
// 2) Row real-FFT (W=256) -> 129 complex bins, radix-2 DIT in LDS
// ---------------------------------------------------------------------------
__global__ __launch_bounds__(256) void fft_rows_kernel(
    const float* __restrict__ x, float2* __restrict__ rowfft) {
  __shared__ float2 buf[256];
  const int row = blockIdx.x;            // (b*C + c)*H + h, 32768 rows
  const int t = threadIdx.x;
  unsigned rv = __brev((unsigned)t) >> 24;
  buf[rv] = make_float2(x[(size_t)row * WW + t], 0.0f);
  __syncthreads();
  for (int s = 1; s <= 8; ++s) {
    const int m = 1 << s, half = m >> 1;
    if (t < 128) {
      const int grp = t / half, pos = t % half;
      const int i0 = grp * m + pos, i1 = i0 + half;
      float sn, cs;
      __sincosf(-6.283185307179586f * (float)pos / (float)m, &sn, &cs);
      float2 a = buf[i0], b = buf[i1];
      float2 wb = make_float2(cs * b.x - sn * b.y, cs * b.y + sn * b.x);
      buf[i0] = make_float2(a.x + wb.x, a.y + wb.y);
      buf[i1] = make_float2(a.x - wb.x, a.y - wb.y);
    }
    __syncthreads();
  }
  if (t < 129) rowfft[(size_t)row * 129 + t] = buf[t];
}

// ---------------------------------------------------------------------------
// 3) Column FFT (H=128 complex) + magnitude mean over channels (ortho norm)
// ---------------------------------------------------------------------------
__global__ __launch_bounds__(128) void fft_cols_kernel(
    const float2* __restrict__ rowfft, float* __restrict__ fe_mean) {
  __shared__ float2 buf[128];
  const int b = blockIdx.x / 129, wbin = blockIdx.x % 129;
  const int t = threadIdx.x;
  const unsigned rv = __brev((unsigned)t) >> 25;
  float acc = 0.0f;
  for (int c = 0; c < CC; ++c) {
    float2 v = rowfft[((size_t)((b * CC + c) * HH) + t) * 129 + wbin];
    __syncthreads();
    buf[rv] = v;
    __syncthreads();
    for (int s = 1; s <= 7; ++s) {
      const int m = 1 << s, half = m >> 1;
      if (t < 64) {
        const int grp = t / half, pos = t % half;
        const int i0 = grp * m + pos, i1 = i0 + half;
        float sn, cs;
        __sincosf(-6.283185307179586f * (float)pos / (float)m, &sn, &cs);
        float2 a = buf[i0], q = buf[i1];
        float2 wb = make_float2(cs * q.x - sn * q.y, cs * q.y + sn * q.x);
        buf[i0] = make_float2(a.x + wb.x, a.y + wb.y);
        buf[i1] = make_float2(a.x - wb.x, a.y - wb.y);
      }
      __syncthreads();
    }
    acc += sqrtf(buf[t].x * buf[t].x + buf[t].y * buf[t].y);
  }
  const float scale = (1.0f / (float)CC) * rsqrtf((float)(HH * WW)); // mean * ortho
  fe_mean[(size_t)(b * HH + t) * 129 + wbin] = acc * scale;
}

// ---------------------------------------------------------------------------
// 4) Build NHWC bf16 aux (144ch) and NHWC bf16 copy of x.
// ---------------------------------------------------------------------------
__global__ __launch_bounds__(256) void aux_build_kernel(
    const float* __restrict__ x, const float* __restrict__ fe_mean,
    unsigned short* __restrict__ auxN, unsigned short* __restrict__ xN) {
  int idx = blockIdx.x * 256 + threadIdx.x;   // 0..65535
  int b = idx >> 15, hw = idx & (HWX - 1);
  int h = hw >> 8, w = hw & 255;
  unsigned short* arow = auxN + (size_t)idx * AUXP;
  unsigned short* xrow = xN + (size_t)idx * CC;
  const float* xp = x + (size_t)b * CC * HWX + hw;
  for (int c = 0; c < CC; ++c) {
    unsigned short v = f2bf(xp[(size_t)c * HWX]);
    arow[c] = v; xrow[c] = v;
  }
  const float PI = 3.14159265358979323846f;
  float theta = -PI + (float)w * (2.0f * PI / 255.0f);
  float phi   = -0.5f * PI + (float)h * (PI / 127.0f);
  float st, ct, sp, cp;
  __sincosf(theta, &st, &ct);
  __sincosf(phi, &sp, &cp);
  arow[128] = f2bf(st); arow[129] = f2bf(ct);
  arow[130] = f2bf(sp); arow[131] = f2bf(cp);
  float sx = ((float)w + 0.5f) * (129.0f / 256.0f) - 0.5f;
  float fx = floorf(sx);
  int x0 = (int)fx;
  float tfrac = sx - fx;
  int x0c = x0 < 0 ? 0 : (x0 > 128 ? 128 : x0);
  int x1c = (x0 + 1) < 0 ? 0 : ((x0 + 1) > 128 ? 128 : (x0 + 1));
  const float* fr = fe_mean + (size_t)(b * HH + h) * 129;
  arow[132] = f2bf((1.0f - tfrac) * fr[x0c] + tfrac * fr[x1c]);
  for (int c = AUXC + 1; c < AUXP; ++c) arow[c] = 0;
}

// ---------------------------------------------------------------------------
// 5) Offset + gate conv: implicit GEMM, M=96 (6 waves), N=16 px, K=1312.
//    Interior tiles pull the raw aux tile via the Tensor Data Mover.
// ---------------------------------------------------------------------------
__global__ __launch_bounds__(192) void conv_og_kernel(
    const unsigned short* __restrict__ auxN, const unsigned short* __restrict__ Wog,
    const float* __restrict__ off_b, const float* __restrict__ g_b1,
    const float* __restrict__ gw2, const float* __restrict__ g_b2,
    float* __restrict__ offset_out, float* __restrict__ gate_out) {
  __shared__ __align__(16) unsigned short Ar[OG_SZ];  // raw tile [row3][px18][144]
  __shared__ __align__(16) unsigned short zpad[8];    // zeros for K tail 1296..1311
  __shared__ float sgate[16];
  const int tile = blockIdx.x;                 // 4096 tiles
  const int b = tile / (HH * 16);
  const int rem = tile % (HH * 16);
  const int h = rem / 16, w0 = (rem % 16) * 16;
  const int tid = threadIdx.x;
  if (tid < 16) sgate[tid] = 0.0f;
  if (tid < 8) zpad[tid] = 0;
  const bool interior = (h >= 1) & (h <= HH - 2) & (w0 >= 16) & (w0 <= WW - 32);
#if USE_TDM
  if (interior) {
    if (tid < 32) {   // one TDM issue per workgroup (wave 0)
      const void* g = auxN + ((size_t)((b * HH + (h - 1)) * WW + (w0 - 1))) * AUXP;
      tdm_load_2d((unsigned)(uintptr_t)&Ar[0], g, 18 * AUXP, 3,
                  (unsigned long)WW * AUXP, false);
      __builtin_amdgcn_s_wait_tensorcnt(0);
    }
  } else
#endif
  {
    for (int i = tid; i < OG_SZ; i += 192) {
      int row = i / (18 * AUXP), r2 = i - row * (18 * AUXP);
      int px = r2 / AUXP, c = r2 - px * AUXP;
      int hh = h + row - 1, ww = w0 + px - 1;
      unsigned short v = 0;
      if (hh >= 0 && hh < HH && ww >= 0 && ww < WW)
        v = auxN[((size_t)((b * HH + hh) * WW + ww)) * AUXP + c];
      Ar[i] = v;
    }
  }
  __syncthreads();
  const int wave = tid >> 5, lane = tid & 31;
  const int pix = lane & 15;
  const int hi8 = (lane & 16) ? 8 : 0;
  const unsigned short* pbase = Ar + pix * AUXP;   // hoist pixel part of B address
  f32x8 acc = {};
#pragma unroll
  for (int kc = 0; kc < KCOG; ++kc) {
    if (kc + 1 < KCOG)
      __builtin_prefetch((const void*)((const uint4*)Wog + (size_t)(wave * KCOG + kc + 1) * 64 + lane * 2), 0, 1);
    const uint4* q1 = og_run(pbase, zpad, kc * 32 + hi8);
    const uint4* q2 = og_run(pbase, zpad, kc * 32 + 16 + hi8);
    bf16x16 bm = frag_from(q1[0], q2[0]);
    bf16x16 a = load_a_frag(Wog, wave * KCOG + kc, lane);
    acc = __builtin_amdgcn_wmma_f32_16x16x32_bf16(false, a, false, bm, (short)0, acc, false, false);
  }
  // epilogue: offsets + fused gate head
  float gpart = 0.0f;
#pragma unroll
  for (int v = 0; v < 8; ++v) {
    int m = wave * 16 + v + hi8;
    float d = acc[v];
    if (m < 18) {
      offset_out[((size_t)(b * 18 + m) * HH + h) * WW + (w0 + pix)] = d + off_b[m];
    } else if (m < 84) {
      float g = d + g_b1[m - 18];
      g = g > 0.0f ? g : 0.0f;                 // relu
      gpart += g * gw2[m - 18];                // fused 1x1 gate conv
    }
  }
  atomicAdd(&sgate[pix], gpart);
  __syncthreads();
  if (tid < 16) {
    float s = sgate[tid] + g_b2[0];
    gate_out[(size_t)(b * HH + h) * WW + w0 + tid] = 1.0f / (1.0f + __expf(-s));
  }
}

// ---------------------------------------------------------------------------
// 6) Fused base conv + deformable sampling + def 3x3 conv + gated blend.
//    Interior tiles pull the base x-tile via TDM (with LDS bank padding);
//    deform patch is gathered concurrently (x is L2-resident).
// ---------------------------------------------------------------------------
__global__ __launch_bounds__(256) void fuse_base_def_kernel(
    const float* __restrict__ x, const unsigned short* __restrict__ xN,
    const unsigned short* __restrict__ Wb, const unsigned short* __restrict__ Wd,
    const float* __restrict__ offset, const float* __restrict__ gate,
    const float* __restrict__ base_b, const float* __restrict__ def_b,
    float* __restrict__ out) {
  __shared__ __align__(16) unsigned short Sp[SP_SZ];     // raw base tile (padded stride)
  __shared__ __align__(16) unsigned short Dp[16 * KB];   // deform-sampled patch
  const int tile = blockIdx.x;
  const int b = tile / (HH * 16);
  const int rem = tile % (HH * 16);
  const int h = rem / 16, w0 = (rem % 16) * 16;
  const int tid = threadIdx.x;
  const bool interior = (h >= 1) & (h <= HH - 2) & (w0 >= 16) & (w0 <= WW - 32);
#if USE_TDM
  if (interior) {
    if (tid < 32) {   // issue DMA early; it overlaps the gather loop below
      const void* g = xN + ((size_t)((b * HH + (h - 1)) * WW + (w0 - 1))) * CC;
      tdm_load_2d((unsigned)(uintptr_t)&Sp[0], g, 18 * CC, 3,
                  (unsigned long)WW * CC, true);
    }
  } else
#endif
  {
    for (int i = tid; i < 3 * 18 * CC; i += 256) {
      int row = i / (18 * CC), r2 = i - row * (18 * CC);
      int px = r2 >> 7, c = r2 & 127;
      int hh = h + row - 1, ww = w0 + px - 1;
      unsigned short v = 0;
      if (hh >= 0 && hh < HH && ww >= 0 && ww < WW)
        v = xN[((size_t)((b * HH + hh) * WW + ww)) * CC + c];
      Sp[((row * 18) + px) * SP_PXS + c] = v;
    }
  }
  // deform patch: k = cin*9 + tap, bilinear gather (L2-resident x)
  for (int i = tid; i < 16 * KB; i += 256) {
    int p = i / KB, k = i - p * KB;
    int c = k / 9, tap = k - c * 9;
    int w = w0 + p;
    size_t ob = ((size_t)(b * 18 + tap * 2) * HH + h) * WW + w;
    float px = (float)w + offset[ob];
    float py = (float)h + offset[ob + (size_t)HWX];
    float fx = floorf(px), fy = floorf(py);
    int x0 = (int)fx, y0 = (int)fy;
    float wx1 = px - fx, wy1 = py - fy;
    float wx0 = 1.0f - wx1, wy0 = 1.0f - wy1;
    const float* xc = x + (size_t)(b * CC + c) * HWX;
    int x1 = x0 + 1, y1 = y0 + 1;
    bool bx0 = (x0 >= 0) & (x0 < WW), bx1 = (x1 >= 0) & (x1 < WW);
    float v00 = 0, v10 = 0, v01 = 0, v11 = 0;
    if (y0 >= 0 && y0 < HH) {
      const float* r0 = xc + (size_t)y0 * WW;
      if (bx0) v00 = r0[x0];
      if (bx1) v10 = r0[x1];
    }
    if (y1 >= 0 && y1 < HH) {
      const float* r1 = xc + (size_t)y1 * WW;
      if (bx0) v01 = r1[x0];
      if (bx1) v11 = r1[x1];
    }
    float s = wy0 * (wx0 * v00 + wx1 * v10) + wy1 * (wx0 * v01 + wx1 * v11);
    Dp[p * KB + k] = f2bf(s);
  }
#if USE_TDM
  if (interior && tid < 32) __builtin_amdgcn_s_wait_tensorcnt(0);
#endif
  __syncthreads();
  const int wave = tid >> 5, lane = tid & 31;
  const int pix = lane & 15;
  const int hi8 = (lane & 16) ? 8 : 0;
  const unsigned short* sbase = Sp + pix * SP_PXS;  // hoist pixel part of B address
  const unsigned short* drow  = Dp + pix * KB;
  f32x8 accB = {}, accD = {};
#pragma unroll
  for (int kc = 0; kc < KCB; ++kc) {
    if (kc + 1 < KCB) {
      __builtin_prefetch((const void*)((const uint4*)Wb + (size_t)(wave * KCB + kc + 1) * 64 + lane * 2), 0, 1);
      __builtin_prefetch((const void*)((const uint4*)Wd + (size_t)(wave * KCB + kc + 1) * 64 + lane * 2), 0, 1);
    }
    const uint4* s1 = sp_run(sbase, kc * 32 + hi8);
    const uint4* s2 = sp_run(sbase, kc * 32 + 16 + hi8);
    bf16x16 ab = load_a_frag(Wb, wave * KCB + kc, lane);
    bf16x16 bb = frag_from(s1[0], s2[0]);
    accB = __builtin_amdgcn_wmma_f32_16x16x32_bf16(false, ab, false, bb, (short)0, accB, false, false);
    bf16x16 ad = load_a_frag(Wd, wave * KCB + kc, lane);
    bf16x16 bd = load_b_frag(drow, kc * 32, lane);
    accD = __builtin_amdgcn_wmma_f32_16x16x32_bf16(false, ad, false, bd, (short)0, accD, false, false);
  }
  const int w = w0 + pix;
  float g = gate[(size_t)(b * HH + h) * WW + w];
#pragma unroll
  for (int v = 0; v < 8; ++v) {
    int m = wave * 16 + v + hi8;
    float yb = accB[v] + base_b[m];
    float yd = accD[v] + def_b[m];
    out[((size_t)(b * CC + m) * HH + h) * WW + w] = (1.0f - g) * yb + g * yd;
  }
}

// ---------------------------------------------------------------------------
// Launch
// ---------------------------------------------------------------------------
extern "C" void kernel_launch(void* const* d_in, const int* in_sizes, int n_in,
                              void* d_out, int out_size, void* d_ws, size_t ws_size,
                              hipStream_t stream) {
  (void)in_sizes; (void)n_in; (void)out_size; (void)ws_size;
  const float* x        = (const float*)d_in[0];
  const float* offset_w = (const float*)d_in[1];
  const float* offset_b = (const float*)d_in[2];
  const float* gate_w1  = (const float*)d_in[3];
  const float* gate_b1  = (const float*)d_in[4];
  const float* gate_w2  = (const float*)d_in[5];
  const float* gate_b2  = (const float*)d_in[6];
  const float* base_w   = (const float*)d_in[7];
  const float* base_b   = (const float*)d_in[8];
  const float* def_w    = (const float*)d_in[9];
  const float* def_b    = (const float*)d_in[10];
  float* out = (float*)d_out;

  char* ws = (char*)d_ws;
  size_t o = 0;
  auto alloc = [&](size_t bytes) { size_t r = o; o = (o + bytes + 255) & ~(size_t)255; return r; };
  float2*         rowfft  = (float2*)(ws + alloc((size_t)BB * CC * HH * 129 * sizeof(float2)));
  float*          fe_mean = (float*)(ws + alloc((size_t)BB * HH * 129 * sizeof(float)));
  unsigned short* auxN    = (unsigned short*)(ws + alloc((size_t)BB * HWX * AUXP * 2));
  unsigned short* xN      = (unsigned short*)(ws + alloc((size_t)BB * HWX * CC * 2));
  unsigned short* Wog     = (unsigned short*)(ws + alloc((size_t)MT_OG * KCOG * 512 * 2));
  unsigned short* Wb      = (unsigned short*)(ws + alloc((size_t)MT_C * KCB * 512 * 2));
  unsigned short* Wd      = (unsigned short*)(ws + alloc((size_t)MT_C * KCB * 512 * 2));
  float*          offs    = (float*)(ws + alloc((size_t)BB * 18 * HWX * sizeof(float)));
  float*          gatev   = (float*)(ws + alloc((size_t)BB * HWX * sizeof(float)));

  const int NPACK = MT_OG * KCOG * 512 + 2 * MT_C * KCB * 512;  // 420864
  pack_weights_kernel<<<(NPACK + 255) / 256, 256, 0, stream>>>(
      offset_w, gate_w1, base_w, def_w, Wog, Wb, Wd);
  fft_rows_kernel<<<BB * CC * HH, 256, 0, stream>>>(x, rowfft);
  fft_cols_kernel<<<BB * 129, 128, 0, stream>>>(rowfft, fe_mean);
  aux_build_kernel<<<(BB * HWX) / 256, 256, 0, stream>>>(x, fe_mean, auxN, xN);
  conv_og_kernel<<<BB * HH * (WW / 16), 192, 0, stream>>>(
      auxN, Wog, offset_b, gate_b1, gate_w2, gate_b2, offs, gatev);
  fuse_base_def_kernel<<<BB * HH * (WW / 16), 256, 0, stream>>>(
      x, xN, Wb, Wd, offs, gatev, base_b, def_b, out);
}
